// S4_4870492914172
// MI455X (gfx1250) — compile-verified, matrix operand order checked by reading, not run on hardware
//
#include <hip/hip_runtime.h>

#define D_MODEL 512
#define D_STATE 16
#define BATCH   4
#define SEQLEN  4096
#define T       32                 // chunk length (= WMMA K dim)
#define NCHUNK  (SEQLEN / T)       // 128
#define CTILES  (NCHUNK / 16)      // 8 column tiles of 16 chunks

typedef __attribute__((ext_vector_type(16))) _Float16 v16h;
typedef __attribute__((ext_vector_type(8)))  float    v8f;

// Load a 16x32 (A) or 32x16 (B, stored K-contiguous per column) f16 operand
// from LDS into the ISA 7.12.2 register layout:
// lane%16 selects row(M)/col(N); hi half of wave offsets K by 8;
// VGPRs 0..3 hold K=0..7(+8*hi), VGPRs 4..7 hold K=16..23(+8*hi).
__device__ __forceinline__ v16h ld_tile(const _Float16* S, int ld, int major0, int lane) {
  const _Float16* rp = S + (size_t)(major0 + (lane & 15)) * ld;
  const int hi = (lane >> 4) & 1;
  v16h x;
#pragma unroll
  for (int v = 0; v < 8; ++v) {
    const int k = ((v < 4) ? (2 * v) : (16 + 2 * (v - 4))) + 8 * hi;
    x[2 * v]     = rp[k];
    x[2 * v + 1] = rp[k + 1];
  }
  return x;
}

// GEMM2 tile: S[reim] = W[reim] @ U   (job is an SGPR value -> uniform flow)
__device__ __forceinline__ void gemm2_job(int job, int lane,
                                          const _Float16* sWre, const _Float16* sWim,
                                          const _Float16* sU, float* sSre, float* sSim) {
  const int reim = job & 1;
  const int ct   = job >> 1;
  v16h aW = ld_tile(reim ? sWim : sWre, T, 0, lane);
  v16h bU = ld_tile(sU, T, ct * 16, lane);
  v8f acc = {};
  acc = __builtin_amdgcn_wmma_f32_16x16x32_f16(false, aW, false, bU,
                                               (short)0, acc, false, false);
  float* Sdst = reim ? sSim : sSre;
  const int nrow = ((lane >> 4) & 1) * 8;
  const int ccol = ct * 16 + (lane & 15);
#pragma unroll
  for (int r = 0; r < 8; ++r)
    Sdst[(nrow + r) * NCHUNK + ccol] = acc[r];
}

// Fused output tile: Y = Toep @ U + [P_re|-P_im] @ [H_re;H_im]  (2 chained WMMAs)
__device__ __forceinline__ void gemm13_job(int job, int lane,
                                           const _Float16* sToep, const _Float16* sPW,
                                           const _Float16* sU, const _Float16* sHp,
                                           float* yout) {
  const int mt = job & 1;
  const int ct = job >> 1;
  v16h aT = ld_tile(sToep, T, mt * 16, lane);
  v16h bU = ld_tile(sU, T, ct * 16, lane);
  v8f acc = {};
  acc = __builtin_amdgcn_wmma_f32_16x16x32_f16(false, aT, false, bU,
                                               (short)0, acc, false, false);
  v16h aP = ld_tile(sPW, T, mt * 16, lane);
  v16h bH = ld_tile(sHp, T, ct * 16, lane);
  acc = __builtin_amdgcn_wmma_f32_16x16x32_f16(false, aP, false, bH,
                                               (short)0, acc, false, false);
  const int mrow = mt * 16 + ((lane >> 4) & 1) * 8;
  const int ccol = ct * 16 + (lane & 15);
#pragma unroll
  for (int r = 0; r < 8; ++r) {
    const int l = ccol * T + mrow + r;
    yout[(size_t)l * D_MODEL] = acc[r];
  }
}

__global__ __launch_bounds__(256) void s4_ssd_kernel(
    const float* __restrict__ u,  const float* __restrict__ Ar,
    const float* __restrict__ Ai, const float* __restrict__ Bm,
    const float* __restrict__ Cm, const float* __restrict__ dl,
    float* __restrict__ y)
{
  __shared__ __align__(16) float    sUf32[SEQLEN];     // async-DMA landing buffer
  __shared__ __align__(16) _Float16 sU[SEQLEN];        // [chunk(128)][j(32)] u in f16
  __shared__ __align__(16) _Float16 sToep[T * T];      // [l][j] = K[l-j]
  __shared__ __align__(16) _Float16 sPW[T * T];        // [l][k] = [P_re | -P_im]
  __shared__ __align__(16) _Float16 sWre[D_STATE * T]; // [n][j] = Re(Abar^{31-j} Bbar)
  __shared__ __align__(16) _Float16 sWim[D_STATE * T];
  __shared__ __align__(16) _Float16 sHp[NCHUNK * T];   // [c][k] = [H_re(k<16) ; H_im]
  __shared__ __align__(16) float sSre[D_STATE * NCHUNK];
  __shared__ __align__(16) float sSim[D_STATE * NCHUNK];
  __shared__ float s_dtAr[16], s_dtAi[16], s_Bbr[16], s_Bbi[16];
  __shared__ float s_CBr[16], s_CBi[16], s_Cv[16], s_Er[16], s_Ei[16];
  __shared__ float s_K[T];

  const int tid  = threadIdx.x;
  const int lane = tid & 31;
  const int wvu  = __builtin_amdgcn_readfirstlane(tid >> 5);  // SGPR wave id
  const int d = blockIdx.x & (D_MODEL - 1);
  const int b = blockIdx.x >> 9;

  // Kick off the only bulk HBM read as async copies into LDS (ASYNCcnt-tracked,
  // no VGPR landing) so the strided gather overlaps the discretization VALU.
  {
    const float* up = u + ((size_t)b * SEQLEN) * D_MODEL + d;
    const unsigned lbase = (unsigned)(uintptr_t)(&sUf32[0]);
    for (int l = tid; l < SEQLEN; l += 256) {
      const unsigned lds_a = lbase + (unsigned)l * 4u;
      const unsigned long long ga =
          (unsigned long long)(uintptr_t)(up + (size_t)l * D_MODEL);
      asm volatile("global_load_async_to_lds_b32 %0, %1, off"
                   :: "v"(lds_a), "v"(ga) : "memory");
    }
  }

  // Phase A.1: ZOH discretization per state n (overlaps the async DMA).
  if (tid < D_STATE) {
    const int n = tid;
    const float ar = Ar[d * D_STATE + n], ai = Ai[d * D_STATE + n];
    const float dt = dl[d];
    const float xr = dt * ar, xi = dt * ai;                   // dtA
    const float e1 = __expf(xr), c1 = __cosf(xi), s1 = __sinf(xi);
    const float abr = e1 * c1 - 1.0f, abi = e1 * s1;          // Abar - 1
    const float den = ar * ar + ai * ai;
    const float inr = ar / den, ini = -ai / den;              // 1/A
    const float Bv = Bm[d * D_STATE + n];
    const float bbr = (abr * inr - abi * ini) * Bv;           // Bbar
    const float bbi = (abr * ini + abi * inr) * Bv;
    const float Cv = Cm[d * D_STATE + n];
    s_dtAr[n] = xr;  s_dtAi[n] = xi;
    s_Bbr[n] = bbr;  s_Bbi[n] = bbi;
    s_CBr[n] = Cv * bbr; s_CBi[n] = Cv * bbi;
    s_Cv[n] = Cv;
    const float eT = __expf((float)T * xr);
    s_Er[n] = eT * __cosf((float)T * xi);                     // E = Abar^T
    s_Ei[n] = eT * __sinf((float)T * xi);
  }

  asm volatile("s_wait_asynccnt 0x0" ::: "memory");
  __syncthreads();   // async data + A.1 results visible to all waves

  // f32 -> f16 conversion of the staged u (LDS -> LDS).
  for (int l = tid; l < SEQLEN; l += 256)
    sU[l] = (_Float16)sUf32[l];

  // Phase A.2: kernel K[k], projection rows P, input rows W (hardware trans ops).
  if (tid < T) {
    const int k = tid;
    float acc = 0.f;
#pragma unroll
    for (int n = 0; n < D_STATE; ++n) {
      const float e = __expf((float)k * s_dtAr[n]);
      const float c = __cosf((float)k * s_dtAi[n]);
      const float s = __sinf((float)k * s_dtAi[n]);
      acc += e * (c * s_CBr[n] - s * s_CBi[n]);               // Re(CB * Abar^k)
    }
    s_K[k] = acc;
    const int l = tid;                                        // P[l][n] = C_n Abar^{l+1}
#pragma unroll
    for (int n = 0; n < D_STATE; ++n) {
      const float ph = (float)(l + 1);
      const float e = __expf(ph * s_dtAr[n]);
      const float c = __cosf(ph * s_dtAi[n]);
      const float s = __sinf(ph * s_dtAi[n]);
      sPW[l * T + n]           = (_Float16)( s_Cv[n] * e * c);
      sPW[l * T + D_STATE + n] = (_Float16)(-s_Cv[n] * e * s);
    }
  }
  if (tid < D_STATE) {
    const int n = tid;                                        // W[n][j] = Abar^{31-j} Bbar
#pragma unroll
    for (int j = 0; j < T; ++j) {
      const float p = (float)(T - 1 - j);
      const float e = __expf(p * s_dtAr[n]);
      const float c = __cosf(p * s_dtAi[n]);
      const float s = __sinf(p * s_dtAi[n]);
      sWre[n * T + j] = (_Float16)(e * (c * s_Bbr[n] - s * s_Bbi[n]));
      sWim[n * T + j] = (_Float16)(e * (s * s_Bbr[n] + c * s_Bbi[n]));
    }
  }
  __syncthreads();

  // Toeplitz fill (needs s_K); GEMM2 below does not touch sToep, so no barrier.
  if (tid < T) {
    const int l = tid;
    for (int j = 0; j < T; ++j)
      sToep[l * T + j] = (_Float16)((j <= l) ? s_K[l - j] : 0.0f);
  }

  // GEMM2: S = W @ U. 16 tiles over 8 waves; job ids are SGPRs -> EXEC stays all-1s.
  gemm2_job(wvu,     lane, sWre, sWim, sU, sSre, sSim);
  gemm2_job(wvu + 8, lane, sWre, sWim, sU, sSre, sSim);
  __syncthreads();

  // Tiny sequential inter-chunk scan: h_{c+1} = E*h_c + S_c (16 complex states).
  // sHp[c] stores the state *entering* chunk c.
  if (tid < D_STATE) {
    const int n = tid;
    float hre = 0.f, him = 0.f;
    const float er = s_Er[n], ei = s_Ei[n];
    for (int c = 0; c < NCHUNK; ++c) {
      sHp[c * T + n]           = (_Float16)hre;
      sHp[c * T + D_STATE + n] = (_Float16)him;
      const float sre = sSre[n * NCHUNK + c];
      const float sim = sSim[n * NCHUNK + c];
      const float t = er * hre - ei * him + sre;
      him = er * him + ei * hre + sim;
      hre = t;
    }
  }
  __syncthreads();

  // Fused GEMM1+GEMM3, straight-line per-wave schedule.
  float* yout = y + ((size_t)b * SEQLEN) * D_MODEL + d;
  gemm13_job(wvu,     lane, sToep, sPW, sU, sHp, yout);
  gemm13_job(wvu + 8, lane, sToep, sPW, sU, sHp, yout);
}

extern "C" void kernel_launch(void* const* d_in, const int* in_sizes, int n_in,
                              void* d_out, int out_size, void* d_ws, size_t ws_size,
                              hipStream_t stream) {
  (void)in_sizes; (void)n_in; (void)out_size; (void)d_ws; (void)ws_size;
  const float* u  = (const float*)d_in[0];
  const float* Ar = (const float*)d_in[1];
  const float* Ai = (const float*)d_in[2];
  const float* Bm = (const float*)d_in[3];
  const float* Cm = (const float*)d_in[4];
  const float* dl = (const float*)d_in[5];
  float* y = (float*)d_out;
  // one workgroup per (batch, channel): 4*512 = 2048 WGs, 8 waves each
  s4_ssd_kernel<<<dim3(BATCH * D_MODEL), dim3(256), 0, stream>>>(u, Ar, Ai, Bm, Cm, dl, y);
}
